// SIRENAutodecoder_mdf_film_14972255994077
// MI455X (gfx1250) — compile-verified, matrix operand order checked by reading, not run on hardware
//
#include <hip/hip_runtime.h>

// ---------------- types ----------------
typedef __bf16 bf16;
typedef __attribute__((ext_vector_type(8)))  bf16  bf16x8;
typedef __attribute__((ext_vector_type(16))) bf16  bf16x16;
typedef __attribute__((ext_vector_type(4)))  float f32x4;
typedef __attribute__((ext_vector_type(8)))  float f32x8;
typedef __attribute__((ext_vector_type(4)))  unsigned int u32x4;
typedef __attribute__((ext_vector_type(8)))  int   i32x8;
typedef __attribute__((ext_vector_type(4)))  int   i32x4;

#define T_    32
#define N_    8192
#define H_    256
#define L_    256
#define NHID_ 4
#define W0_   30.0f

// =====================================================================
// Kernel 1: W_eff[l][t][o][k] = nf_w_mid[l][o][k] + sum_j hw_w_mid[l][o*256+k][j] * lat[t][j]
// One wave handles 32 consecutive rows r (r = o*256+k); lane owns one row,
// streams it once from HBM (268MB total read once), keeps 32 per-scene
// accumulators in registers.  Writes are lane-coalesced (64B / scene).
// =====================================================================
__global__ __launch_bounds__(256) void build_wmid(
    const float* __restrict__ latents,
    const float* __restrict__ nf_w_mid,
    const float* __restrict__ hw_w_mid,
    bf16* __restrict__ wmid)
{
    __shared__ float lat[32 * 260];                 // padded stride 260
    const int tid = threadIdx.x;
    for (int idx = tid; idx < 32 * 256; idx += 256) {
        int t = idx >> 8, j = idx & 255;
        lat[t * 260 + j] = latents[t * 256 + j];
    }
    __syncthreads();

    const int wave = tid >> 5, lane = tid & 31;
    const int g  = blockIdx.x * 8 + wave;           // 0..8191 (one 32-row chunk each)
    const int r0 = g << 5;                          // global row base (0..262143)
    const int l  = r0 >> 16;                        // layer 0..3
    const int row = (r0 & 65535) + lane;            // this lane's row within layer

    const float* src = hw_w_mid + (((size_t)l << 16) + row) * 256;
    const float  base = nf_w_mid[((size_t)l << 16) + row];

    float acc[32];
#pragma unroll
    for (int t = 0; t < 32; ++t) acc[t] = base;

    for (int j = 0; j < 256; j += 4) {
        f32x4 h = *(const f32x4*)(src + j);
#pragma unroll
        for (int t = 0; t < 32; ++t) {
            f32x4 lt = *(const f32x4*)(lat + t * 260 + j);  // LDS broadcast
            acc[t] += h.x * lt.x + h.y * lt.y + h.z * lt.z + h.w * lt.w;
        }
    }
#pragma unroll
    for (int t = 0; t < 32; ++t)
        wmid[(((size_t)(l * 32 + t)) << 16) + row] = (bf16)acc[t];
}

// =====================================================================
// Kernel 2: tiny effective params (first-layer weights/bias, mid biases).
// job = rowid*32 + t; each wave shares one source row (uniform -> scalarized),
// lanes differ only in scene t.
// =====================================================================
__global__ __launch_bounds__(256) void build_small(
    const float* __restrict__ latents,
    const float* __restrict__ nf_w_first,
    const float* __restrict__ nf_b_first,
    const float* __restrict__ nf_b_mid,
    const float* __restrict__ hw_w_first,
    const float* __restrict__ hb_w,
    float* __restrict__ wfirst,    // [t][512]
    float* __restrict__ bfirst,    // [t][256]
    float* __restrict__ bmid)      // [l][t][256]
{
    __shared__ float lat[32 * 260];
    const int tid = threadIdx.x;
    for (int idx = tid; idx < 32 * 256; idx += 256) {
        int t = idx >> 8, j = idx & 255;
        lat[t * 260 + j] = latents[t * 256 + j];
    }
    __syncthreads();

    const int job = blockIdx.x * 256 + tid;         // < 1792*32
    const int rowid = job >> 5;
    const int t = job & 31;

    const float* src;
    float addv;
    float* dst;
    if (rowid < 512) {                              // first-layer weights (o*2+c)
        src  = hw_w_first + (size_t)rowid * 256;
        addv = nf_w_first[rowid];
        dst  = wfirst + t * 512 + rowid;
    } else if (rowid < 768) {                       // first-layer bias (hb_w[0])
        int o = rowid - 512;
        src  = hb_w + (size_t)o * 256;
        addv = nf_b_first[o];
        dst  = bfirst + t * 256 + o;
    } else {                                        // mid biases (hb_w[1..4])
        int q = rowid - 768, l = q >> 8, o = q & 255;
        src  = hb_w + ((size_t)(l + 1) * 256 + o) * 256;
        addv = nf_b_mid[l * 256 + o];
        dst  = bmid + ((l * 32 + t) * 256) + o;
    }

    float acc = addv;
    for (int j = 0; j < 256; j += 4) {
        f32x4 h  = *(const f32x4*)(src + j);        // wave-uniform
        f32x4 lt = *(const f32x4*)(lat + t * 260 + j);
        acc += h.x * lt.x + h.y * lt.y + h.z * lt.z + h.w * lt.w;
    }
    *dst = acc;
}

// =====================================================================
// Fragment loader (A and B use the same per-lane pattern):
//   lane L<16 : row = rbase+L,    K = {k0..k0+7, k0+16..k0+23}
//   lane L>=16: row = rbase+L-16, K = {k0+8..k0+15, k0+24..k0+31}
// rows have padded stride 264 halfs in LDS (bank-conflict mitigation).
// =====================================================================
__device__ __forceinline__ bf16x16 frag16(const bf16* base, int rbase, int k0, int lane)
{
    const int r = lane & 15;
    const int h = (lane >> 4) & 1;
    const bf16* p = base + (rbase + r) * 264 + k0 + h * 8;
    union { bf16x16 v; bf16x8 half2[2]; } u;
    u.half2[0] = *(const bf16x8*)p;
    u.half2[1] = *(const bf16x8*)(p + 16);
    return u.v;
}

// =====================================================================
// TDM: DMA one 256x256 bf16 W_eff tile from global into LDS.
// D# pad fields insert 16B of padding after every 512B row, producing the
// 264-half LDS row stride the WMMA fragment loader expects.
// Issued by wave 0 only (scalar-uniform branch); tracked with TENSORcnt.
// =====================================================================
__device__ __forceinline__ void tdm_load_w(const bf16* gsrc, bf16* ldst)
{
    unsigned long long ga = (unsigned long long)gsrc;
    unsigned int lds = (unsigned int)(unsigned long long)ldst;  // LDS byte address = addr[31:0]

    u32x4 g0;
    g0.x = 1u;                                        // count=1 (valid user descriptor)
    g0.y = lds;                                       // lds_addr
    g0.z = (unsigned int)(ga & 0xFFFFFFFFu);          // global_addr[31:0]
    g0.w = (unsigned int)(ga >> 32) | (2u << 30);     // global_addr[56:32] | type=2

    i32x8 g1;
    g1[0] = (1 << 16)                                 // data_size = 2 bytes
          | (1 << 20)                                 // pad_enable
          | (6 << 22)                                 // pad_interval: 128 DWORDs (512B row)
          | (3 << 25);                                // pad_amount : 4 DWORDs (16B)
    g1[1] = (int)(256u << 16);                        // tensor_dim0 = 256
    g1[2] = (int)(256u << 16);                        // tensor_dim1 = 256 (lo16)
    g1[3] = (int)(256u << 16);                        // tile_dim0 = 256
    g1[4] = 256;                                      // tile_dim1 = 256, tile_dim2 = 0
    g1[5] = 256;                                      // tensor_dim0_stride = 256
    g1[6] = 0;
    g1[7] = 0;

    i32x4 z4 = {0, 0, 0, 0};                          // 2-D tensor: groups 2/3 unused
    i32x8 z8 = {0, 0, 0, 0, 0, 0, 0, 0};              // trailing group (clang-23 6-arg form)
    __builtin_amdgcn_tensor_load_to_lds(g0, g1, z4, z4, z8, 0);
}

// =====================================================================
// Kernel 3: the whole network for one (scene t, 128-row tile).
// x tile lives in LDS across all layers (zero activation HBM traffic).
// Per layer: TDM-stage W_eff(t,l) (128KB bf16) into LDS, 8 waves tiled
// 4(M)x2(N); each wave: 2x8 accumulator tiles, v_wmma_f32_16x16x32_bf16
// over K=256.  sin() writeback overlaps the next layer's TDM DMA.
// =====================================================================
__global__ __launch_bounds__(256) void siren_main(
    const float* __restrict__ coords,
    const float* __restrict__ wfirst,
    const float* __restrict__ bfirst,
    const bf16*  __restrict__ wmid,
    const float* __restrict__ bmid,
    const float* __restrict__ nf_w_last,
    const float* __restrict__ nf_b_last,
    float* __restrict__ out)
{
    extern __shared__ char smem[];
    bf16*  xs   = (bf16*)(smem);                    // 128 x 264 halfs  (67584 B)
    bf16*  ww   = (bf16*)(smem + 67584);            // 256 x 264 halfs  (135168 B)
    bf16*  wl   = (bf16*)(smem + 202752);           // 16  x 264 halfs  (8448 B)
    float* bias = (float*)(smem + 211200);          // 256 f32
    float* cl   = (float*)(smem + 212224);          // 128x2 f32

    const int tid  = threadIdx.x;
    const int lane = tid & 31;
    const int wv   = tid >> 5;                      // 0..7
    const int t    = blockIdx.y;
    const int n0   = blockIdx.x * 128;

    // ---------------- phase 0: stage small params; TDM kicks layer-0 W ----------
    if (__builtin_amdgcn_readfirstlane(tid) == 0)   // wave 0 only (scalar branch)
        tdm_load_w(wmid + ((size_t)t << 16), ww);

    for (int idx = tid; idx < 16 * 264; idx += 256) {
        int rr = idx / 264, kk = idx % 264;
        wl[idx] = (rr < 3 && kk < 256) ? (bf16)nf_w_last[rr * 256 + kk] : (bf16)0.0f;
    }
    cl[tid] = coords[(size_t)t * N_ * 2 + (size_t)n0 * 2 + tid];
    bias[tid] = bmid[t * 256 + tid];
    const float w00 = wfirst[t * 512 + tid * 2];
    const float w01 = wfirst[t * 512 + tid * 2 + 1];
    const float bb  = bfirst[t * 256 + tid];
    __syncthreads();

    // ---------------- first layer (K=2, scalar): write xs; overlaps TDM ----------
    for (int m = 0; m < 128; ++m) {
        float y = fmaf(cl[2 * m], w00, fmaf(cl[2 * m + 1], w01, bb));
        xs[m * 264 + tid] = (bf16)__sinf(W0_ * y);
    }
    if (__builtin_amdgcn_readfirstlane(tid) == 0)   // layer-0 W must have landed
        __builtin_amdgcn_s_wait_tensorcnt(0);
    __syncthreads();

    // ---------------- 4 hidden layers, WMMA ----------
    const int mw = wv & 3;                          // M group: rows [mw*32, +32)
    const int nw = wv >> 2;                         // N group: cols [nw*128, +128)

    for (int l = 0; l < NHID_; ++l) {
        if (l < NHID_ - 1) {                        // warm GL2 for next layer's W_eff
            const char* nxt = (const char*)(wmid + ((size_t)((l + 1) * 32 + t) << 16));
#pragma unroll
            for (int q = 0; q < 8; ++q)
                __builtin_prefetch(nxt + (size_t)tid * 512 + q * 64, 0, 1);
        }

        f32x8 acc0[8], acc1[8];
#pragma unroll
        for (int jj = 0; jj < 8; ++jj) {
            float b = bias[nw * 128 + jj * 16 + (lane & 15)];   // bias folded into init
#pragma unroll
            for (int i = 0; i < 8; ++i) { acc0[jj][i] = b; acc1[jj][i] = b; }
        }

        for (int k0 = 0; k0 < 256; k0 += 32) {
            bf16x16 a0 = frag16(xs, mw * 32,      k0, lane);
            bf16x16 a1 = frag16(xs, mw * 32 + 16, k0, lane);
#pragma unroll
            for (int jj = 0; jj < 8; ++jj) {
                bf16x16 bm = frag16(ww, nw * 128 + jj * 16, k0, lane);
                acc0[jj] = __builtin_amdgcn_wmma_f32_16x16x32_bf16(
                    false, a0, false, bm, (short)0, acc0[jj], false, false);
                acc1[jj] = __builtin_amdgcn_wmma_f32_16x16x32_bf16(
                    false, a1, false, bm, (short)0, acc1[jj], false, false);
            }
        }
        __syncthreads();                            // all waves done reading xs/ww

        // kick next layer's W DMA first so it overlaps the sin writeback
        if (l < NHID_ - 1 && __builtin_amdgcn_readfirstlane(tid) == 0)
            tdm_load_w(wmid + ((size_t)((l + 1) * 32 + t) << 16), ww);

        // sin activation, write back in place into xs; stage next bias
        const int rb = mw * 32 + ((lane >> 4) << 3);
#pragma unroll
        for (int jj = 0; jj < 8; ++jj) {
            int col = nw * 128 + jj * 16 + (lane & 15);
#pragma unroll
            for (int i = 0; i < 8; ++i) {
                xs[(rb + i)      * 264 + col] = (bf16)__sinf(W0_ * acc0[jj][i]);
                xs[(rb + 16 + i) * 264 + col] = (bf16)__sinf(W0_ * acc1[jj][i]);
            }
        }
        if (l < NHID_ - 1) {
            bias[tid] = bmid[((l + 1) * 32 + t) * 256 + tid];
            if (__builtin_amdgcn_readfirstlane(tid) == 0)
                __builtin_amdgcn_s_wait_tensorcnt(0);
        }
        __syncthreads();
    }

    // ---------------- last layer (OUT=3) via WMMA against zero-padded B ----------
    {
        f32x8 acc;
#pragma unroll
        for (int i = 0; i < 8; ++i) acc[i] = 0.0f;
        for (int k0 = 0; k0 < 256; k0 += 32) {
            bf16x16 a = frag16(xs, wv * 16, k0, lane);   // wave wv: rows [wv*16,+16)
            bf16x16 b = frag16(wl, 0, k0, lane);
            acc = __builtin_amdgcn_wmma_f32_16x16x32_bf16(
                false, a, false, b, (short)0, acc, false, false);
        }
        const int r = lane & 15;
        if (r < 3) {
            float bl = nf_b_last[r];
            int rb = wv * 16 + ((lane >> 4) << 3);
#pragma unroll
            for (int i = 0; i < 8; ++i)
                out[((size_t)t * N_ + n0 + rb + i) * 3 + r] = acc[i] + bl;
        }
    }
}

// =====================================================================
// launch
// =====================================================================
extern "C" void kernel_launch(void* const* d_in, const int* in_sizes, int n_in,
                              void* d_out, int out_size, void* d_ws, size_t ws_size,
                              hipStream_t stream)
{
    (void)in_sizes; (void)n_in; (void)out_size; (void)ws_size;

    const float* coords     = (const float*)d_in[0];
    const float* latents    = (const float*)d_in[1];
    const float* nf_w_first = (const float*)d_in[2];
    const float* nf_b_first = (const float*)d_in[3];
    const float* nf_w_mid   = (const float*)d_in[4];
    const float* nf_b_mid   = (const float*)d_in[5];
    const float* nf_w_last  = (const float*)d_in[6];
    const float* nf_b_last  = (const float*)d_in[7];
    const float* hw_w_first = (const float*)d_in[8];
    const float* hw_w_mid   = (const float*)d_in[9];
    const float* hb_w       = (const float*)d_in[10];

    // workspace: W_eff bf16 (16.78MB) | bmid | wfirst | bfirst  (~17MB total)
    bf16*  wmid   = (bf16*)d_ws;
    float* bmid   = (float*)((char*)d_ws + (size_t)NHID_ * 32 * 65536 * 2);
    float* wfirst = bmid + NHID_ * 32 * 256;
    float* bfirst = wfirst + 32 * 512;

    build_wmid<<<dim3(1024), dim3(256), 0, stream>>>(latents, nf_w_mid, hw_w_mid, wmid);
    build_small<<<dim3(224), dim3(256), 0, stream>>>(latents, nf_w_first, nf_b_first,
                                                     nf_b_mid, hw_w_first, hb_w,
                                                     wfirst, bfirst, bmid);
    siren_main<<<dim3(64, 32), dim3(256), 213248, stream>>>(coords, wfirst, bfirst,
                                                            wmid, bmid, nf_w_last,
                                                            nf_b_last, (float*)d_out);
}
